// Qwen3_5Attention_41016937677097
// MI455X (gfx1250) — matrix operand order repaired and checked by reading, not compiled, not run on hardware
//
#include <hip/hip_runtime.h>

// ---------------- problem constants ----------------
#define T_    4096
#define H_    2048
#define NH_   16
#define NKV_  2
#define HD_   256
#define G_    (NH_ / NKV_)      // 8
#define QGW   (NH_ * HD_ * 2)   // 8192
#define KVW   (NKV_ * HD_)      // 512
#define OW    (NH_ * HD_)       // 4096
#define SCALE_ 0.0625f          // HD^-0.5 = 1/16
#define EPS_   1e-6f
#define NEG_BIG (-1.0e30f)

// async global->LDS staging via inline asm (typed builtin params are not
// directly reachable from HIP generic pointers)
#define USE_ASYNC_ASM 1

#if defined(__has_builtin) && __has_builtin(__builtin_amdgcn_s_wait_asynccnt)
#define WAIT_ASYNC0() __builtin_amdgcn_s_wait_asynccnt(0)
#else
#define WAIT_ASYNC0() asm volatile("s_wait_asynccnt 0" ::: "memory")
#endif

typedef __attribute__((ext_vector_type(16))) __bf16 v16bf;
typedef __attribute__((ext_vector_type(8)))  __bf16 v8bf;
typedef __attribute__((ext_vector_type(8)))  float  v8f;
typedef __attribute__((ext_vector_type(8)))  unsigned short v8us;

// ---------------- helpers ----------------
static __device__ __forceinline__ unsigned short f2bf(float f) {
    unsigned int u = __float_as_uint(f);
    unsigned int r = u + 0x7FFFu + ((u >> 16) & 1u);   // round-to-nearest-even
    return (unsigned short)(r >> 16);
}
static __device__ __forceinline__ float bf2f(unsigned short u) {
    return __uint_as_float(((unsigned int)u) << 16);
}
static __device__ __forceinline__ v8bf us2bf8(v8us u) {
    v8bf r; __builtin_memcpy(&r, &u, 16); return r;
}
static __device__ __forceinline__ v16bf mk16(v8bf lo, v8bf hi) {
    return __builtin_shufflevector(lo, hi, 0,1,2,3,4,5,6,7,8,9,10,11,12,13,14,15);
}
// one async 16-byte copy: lds_byte_addr <- *gsrc  (offset imm applies to BOTH sides)
static __device__ __forceinline__ void async_b128(unsigned lds_byte_addr,
                                                  const unsigned short* gsrc) {
    asm volatile("global_load_async_to_lds_b128 %0, %1, off"
                 :: "v"(lds_byte_addr), "v"(gsrc) : "memory");
}
static __device__ __forceinline__ void async_b128_off16(unsigned lds_byte_addr,
                                                        const unsigned short* gsrc) {
    asm volatile("global_load_async_to_lds_b128 %0, %1, off offset:16"
                 :: "v"(lds_byte_addr), "v"(gsrc) : "memory");
}
static __device__ __forceinline__ void async_b128_off32(unsigned lds_byte_addr,
                                                        const unsigned short* gsrc) {
    asm volatile("global_load_async_to_lds_b128 %0, %1, off offset:32"
                 :: "v"(lds_byte_addr), "v"(gsrc) : "memory");
}
static __device__ __forceinline__ void async_b128_off48(unsigned lds_byte_addr,
                                                        const unsigned short* gsrc) {
    asm volatile("global_load_async_to_lds_b128 %0, %1, off offset:48"
                 :: "v"(lds_byte_addr), "v"(gsrc) : "memory");
}

// ---------------- f32 -> bf16 streaming convert (row-major) ----------------
__global__ void f32_to_bf16_kernel(const float* __restrict__ s,
                                   unsigned short* __restrict__ d, long long n) {
    long long i = (long long)blockIdx.x * blockDim.x + threadIdx.x;
    long long stride = (long long)gridDim.x * blockDim.x;
    for (; i < n; i += stride) d[i] = f2bf(s[i]);
}

// ---------------- f32 (KxN) -> bf16 transposed (NxK) ----------------
__global__ __launch_bounds__(256)
void transpose_f32_bf16_kernel(const float* __restrict__ src,
                               unsigned short* __restrict__ dst, int K, int N) {
    __shared__ float tile[32][33];
    const int k0 = blockIdx.y * 32, n0 = blockIdx.x * 32;
    for (int i = threadIdx.y; i < 32; i += 8)
        tile[i][threadIdx.x] = src[(size_t)(k0 + i) * N + n0 + threadIdx.x];
    __syncthreads();
    for (int i = threadIdx.y; i < 32; i += 8)
        dst[(size_t)(n0 + i) * K + k0 + threadIdx.x] = f2bf(tile[threadIdx.x][i]);
}

// ---------------- bf16 WMMA GEMM: C(MxN) = A(MxK) * BT(NxK)^T ----------------
// block tile 64x128, 128 threads (4 waves); wave w: rows [16w,16w+16), 8 accums.
// storeMode: 0 = f32 row-major, 1 = bf16 row-major, 2 = bf16 transposed [col][row]
// LDS: single shared array (base offset 0): ldsA = smem[0..2047]   (64 x 32, [m][k])
//                                           ldsB = smem[2048..6143] (128 x 32, [n][k])
__global__ __launch_bounds__(128)
void gemm_bf16_tn_kernel(const unsigned short* __restrict__ A,
                         const unsigned short* __restrict__ BT,
                         void* __restrict__ C,
                         int M, int N, int K, int storeMode) {
    __shared__ unsigned short smem[2048 + 4096];
    // Escape the LDS base address: the async copies below write smem through
    // raw LDS byte addresses inside inline asm, so without this the compiler's
    // alias analysis proves smem is never written and deletes the frag loads.
    asm volatile("" :: "v"((const void*)&smem[0]) : "memory");

    const int tid  = threadIdx.x;
    const int lane = tid & 31;
    const int wave = tid >> 5;
    const int hf   = lane >> 4;
    const int l15  = lane & 15;
    const int rowBase = blockIdx.y * 64;
    const int colBase = blockIdx.x * 128;

    v8f acc[8];
#pragma unroll
    for (int f = 0; f < 8; ++f)
#pragma unroll
        for (int i = 0; i < 8; ++i) acc[f][i] = 0.0f;

    const int ar = tid >> 1, akc = (tid & 1) * 16;
    const unsigned ldsA_addr = (unsigned)((ar * 32 + akc) * 2);       // bytes
    const unsigned ldsB_addr = (unsigned)((2048 + tid * 32) * 2);     // bytes

    for (int kk = 0; kk < K; kk += 32) {
        // ---- stage tiles: both are contiguous 32-elem row segments ----
        const unsigned short* gA = A + (size_t)(rowBase + ar) * K + kk + akc;
        const unsigned short* gB = BT + (size_t)(colBase + tid) * K + kk;
#if USE_ASYNC_ASM
        async_b128(ldsA_addr, gA);
        async_b128_off16(ldsA_addr, gA);
        async_b128(ldsB_addr, gB);
        async_b128_off16(ldsB_addr, gB);
        async_b128_off32(ldsB_addr, gB);
        async_b128_off48(ldsB_addr, gB);
        WAIT_ASYNC0();
#else
        {
            const v8us* sA = (const v8us*)gA;
            v8us a0 = sA[0], a1 = sA[1];
            v8us* dA = (v8us*)&smem[ar * 32 + akc];
            dA[0] = a0; dA[1] = a1;
            const v8us* sB = (const v8us*)gB;
            v8us b0 = sB[0], b1 = sB[1], b2 = sB[2], b3 = sB[3];
            v8us* dB = (v8us*)&smem[2048 + tid * 32];
            dB[0] = b0; dB[1] = b1; dB[2] = b2; dB[3] = b3;
        }
#endif
        __syncthreads();

        // ---- compute ----
        const int arow = wave * 16 + l15;
        v8us alo = *(const v8us*)&smem[arow * 32 + hf * 8];
        v8us ahi = *(const v8us*)&smem[arow * 32 + 16 + hf * 8];
        v16bf afrag = mk16(us2bf8(alo), us2bf8(ahi));
#pragma unroll
        for (int f = 0; f < 8; ++f) {
            const int n = f * 16 + l15;
            v8us blo = *(const v8us*)&smem[2048 + n * 32 + hf * 8];
            v8us bhi = *(const v8us*)&smem[2048 + n * 32 + 16 + hf * 8];
            v16bf bfrag = mk16(us2bf8(blo), us2bf8(bhi));
            acc[f] = __builtin_amdgcn_wmma_f32_16x16x32_bf16(
                false, afrag, false, bfrag, (short)0, acc[f], false, false);
        }
        __syncthreads();
    }

    // ---- epilogue ----
    if (storeMode == 1) {
        unsigned short* Cb = (unsigned short*)C;
#pragma unroll
        for (int f = 0; f < 8; ++f)
#pragma unroll
            for (int r = 0; r < 8; ++r) {
                int row = rowBase + wave * 16 + r + hf * 8;
                int col = colBase + f * 16 + l15;
                Cb[(size_t)row * N + col] = f2bf(acc[f][r]);
            }
    } else if (storeMode == 2) {
        unsigned short* Cb = (unsigned short*)C;   // [col][row], i.e. N x M
#pragma unroll
        for (int f = 0; f < 8; ++f)
#pragma unroll
            for (int r = 0; r < 8; ++r) {
                int row = rowBase + wave * 16 + r + hf * 8;
                int col = colBase + f * 16 + l15;
                Cb[(size_t)col * M + row] = f2bf(acc[f][r]);
            }
    } else {
        float* Cf = (float*)C;
#pragma unroll
        for (int f = 0; f < 8; ++f)
#pragma unroll
            for (int r = 0; r < 8; ++r) {
                int row = rowBase + wave * 16 + r + hf * 8;
                int col = colBase + f * 16 + l15;
                Cf[(size_t)row * N + col] = acc[f][r];
            }
    }
}

// ---------------- RMS norm + RoPE (in place, bf16) ----------------
__global__ __launch_bounds__(256)
void normrope_kernel(unsigned short* __restrict__ data,
                     const float* __restrict__ w,
                     const int* __restrict__ pos,
                     int heads, int rowStride, int headStride) {
    __shared__ float red[256];
    __shared__ float ybuf[256];
    const int t  = blockIdx.x / heads;
    const int hh = blockIdx.x % heads;
    unsigned short* row = data + (size_t)t * rowStride + (size_t)hh * headStride;
    const int d = threadIdx.x;

    float x = bf2f(row[d]);
    red[d] = x * x;
    __syncthreads();
    for (int sft = 128; sft > 0; sft >>= 1) {
        if (d < sft) red[d] += red[d + sft];
        __syncthreads();
    }
    float inv = rsqrtf(red[0] / 256.0f + EPS_);
    float y = x * inv * (1.0f + w[d]);
    ybuf[d] = y;
    __syncthreads();

    float out = y;
    if (d < 64) {
        int i = (d < 32) ? d : d - 32;
        float inv_freq = expf(-((float)(2 * i) / 64.0f) * logf(10000000.0f));
        int axis = i % 3;
        float p  = (float)pos[axis * T_ + t];
        float fr = p * inv_freq;
        float c = cosf(fr), sn = sinf(fr);
        float x1 = ybuf[i], x2 = ybuf[i + 32];
        out = (d < 32) ? (x1 * c - x2 * sn) : (x2 * c + x1 * sn);
    }
    row[d] = f2bf(out);
}

// ---------------- flash attention + gate ----------------
// grid (T/16, NH), 128 threads (4 waves). Wave w owns output dims [64w, 64w+64).
// Vt is dim-major: Vt[(kvh*HD + dim)*T + t]
__global__ __launch_bounds__(128)
void attn_kernel(const unsigned short* __restrict__ QG,
                 const unsigned short* __restrict__ Kb,
                 const unsigned short* __restrict__ Vt,
                 unsigned short* __restrict__ Ob) {
    __shared__ unsigned short ldsP[16][32];    // P tile (row-major)
    const int tid  = threadIdx.x;
    const int lane = tid & 31;
    const int wave = tid >> 5;
    const int hf   = lane >> 4;
    const int l15  = lane & 15;
    const int qb   = blockIdx.x;
    const int h    = blockIdx.y;
    const int kvh  = h / G_;
    const int qrow0 = qb * 16;

    // preload Q fragments (16 rows x 256 dims -> 8 A-frags)
    v16bf qf[8];
    {
        const unsigned short* qrow = QG + (size_t)(qrow0 + l15) * QGW + h * (2 * HD_);
#pragma unroll
        for (int c = 0; c < 8; ++c) {
            v8us lo = *(const v8us*)(qrow + 32 * c + hf * 16);
            v8us hi = *(const v8us*)(qrow + 32 * c + hf * 16 + 8);
            qf[c] = mk16(us2bf8(lo), us2bf8(hi));
        }
    }

    v8f o[4];
#pragma unroll
    for (int f = 0; f < 4; ++f)
#pragma unroll
        for (int i = 0; i < 8; ++i) o[f][i] = 0.0f;
    float mrow[8], lrow[8];
#pragma unroll
    for (int r = 0; r < 8; ++r) { mrow[r] = NEG_BIG; lrow[r] = 0.0f; }

    const int ktmax = qb >> 1;
    for (int kt = 0; kt <= ktmax; ++kt) {
        const int kb = kt * 32;

        // ---- scores S = Q * K^T (16x32), each wave redundantly ----
        v8f s0, s1;
#pragma unroll
        for (int i = 0; i < 8; ++i) { s0[i] = 0.0f; s1[i] = 0.0f; }
        {
            const unsigned short* krow0 = Kb + (size_t)(kb + l15) * KVW + kvh * HD_;
            const unsigned short* krow1 = krow0 + 16 * KVW;
#pragma unroll
            for (int c = 0; c < 8; ++c) {
                v8us a0 = *(const v8us*)(krow0 + 32 * c + hf * 16);
                v8us a1 = *(const v8us*)(krow0 + 32 * c + hf * 16 + 8);
                v16bf kf0 = mk16(us2bf8(a0), us2bf8(a1));
                s0 = __builtin_amdgcn_wmma_f32_16x16x32_bf16(
                    false, qf[c], false, kf0, (short)0, s0, false, false);
                v8us b0 = *(const v8us*)(krow1 + 32 * c + hf * 16);
                v8us b1 = *(const v8us*)(krow1 + 32 * c + hf * 16 + 8);
                v16bf kf1 = mk16(us2bf8(b0), us2bf8(b1));
                s1 = __builtin_amdgcn_wmma_f32_16x16x32_bf16(
                    false, qf[c], false, kf1, (short)0, s1, false, false);
            }
        }

        // ---- mask, online softmax ----
        float p0v[8], p1v[8], alpha[8];
#pragma unroll
        for (int r = 0; r < 8; ++r) {
            int qrow = qrow0 + r + hf * 8;
            int key0 = kb + l15, key1 = key0 + 16;
            float x0 = s0[r] * SCALE_; if (key0 > qrow) x0 = NEG_BIG;
            float x1 = s1[r] * SCALE_; if (key1 > qrow) x1 = NEG_BIG;
            float tm = fmaxf(x0, x1);
            for (int msk = 1; msk < 16; msk <<= 1)
                tm = fmaxf(tm, __shfl_xor(tm, msk, 32));
            float mn = fmaxf(mrow[r], tm);
            float a  = __expf(mrow[r] - mn);
            float p0 = __expf(x0 - mn), p1 = __expf(x1 - mn);
            float rs = p0 + p1;
            for (int msk = 1; msk < 16; msk <<= 1)
                rs += __shfl_xor(rs, msk, 32);
            lrow[r] = lrow[r] * a + rs;
            mrow[r] = mn;
            alpha[r] = a; p0v[r] = p0; p1v[r] = p1;
        }
#pragma unroll
        for (int f = 0; f < 4; ++f)
#pragma unroll
            for (int r = 0; r < 8; ++r) o[f][r] *= alpha[r];

        if (wave == 0) {
#pragma unroll
            for (int r = 0; r < 8; ++r) {
                ldsP[r + hf * 8][l15]      = f2bf(p0v[r]);
                ldsP[r + hf * 8][l15 + 16] = f2bf(p1v[r]);
            }
        }
        __syncthreads();   // P visible

        // ---- O += P * V (wave's 64-dim slice); V read dim-major from global ----
        v16bf pf;
        {
            v8us lo = *(const v8us*)&ldsP[l15][hf * 8];
            v8us hi = *(const v8us*)&ldsP[l15][16 + hf * 8];
            pf = mk16(us2bf8(lo), us2bf8(hi));
        }
#pragma unroll
        for (int f = 0; f < 4; ++f) {
            int dim = wave * 64 + f * 16 + l15;
            const unsigned short* vcol =
                Vt + (size_t)(kvh * HD_ + dim) * T_ + kb + hf * 8;
            v8us lo = *(const v8us*)vcol;
            v8us hi = *(const v8us*)(vcol + 16);
            v16bf vf = mk16(us2bf8(lo), us2bf8(hi));
            o[f] = __builtin_amdgcn_wmma_f32_16x16x32_bf16(
                false, pf, false, vf, (short)0, o[f], false, false);
        }
        __syncthreads();   // safe to overwrite ldsP next iter
    }

    // ---- epilogue: 1/l, sigmoid(gate), store bf16 ----
#pragma unroll
    for (int f = 0; f < 4; ++f)
#pragma unroll
        for (int r = 0; r < 8; ++r) {
            int qrow = qrow0 + r + hf * 8;
            int dim  = wave * 64 + f * 16 + l15;
            float val = o[f][r] / lrow[r];
            float g = bf2f(QG[(size_t)qrow * QGW + h * (2 * HD_) + HD_ + dim]);
            val *= 1.0f / (1.0f + __expf(-g));
            Ob[(size_t)qrow * OW + h * HD_ + dim] = f2bf(val);
        }
}

// ---------------- host launcher ----------------
extern "C" void kernel_launch(void* const* d_in, const int* in_sizes, int n_in,
                              void* d_out, int out_size, void* d_ws, size_t ws_size,
                              hipStream_t stream) {
    const int*   positions = (const int*)d_in[0];
    const float* hidden    = (const float*)d_in[1];
    const float* Wq        = (const float*)d_in[2];
    const float* Wk        = (const float*)d_in[3];
    const float* Wv        = (const float*)d_in[4];
    const float* Wo        = (const float*)d_in[5];
    const float* qw        = (const float*)d_in[6];
    const float* kw        = (const float*)d_in[7];

    char* ws = (char*)d_ws;
    size_t off = 0;
    auto alloc = [&](size_t bytes) -> void* {
        void* p = ws + off;
        off += (bytes + 255) & ~(size_t)255;
        return p;
    };
    unsigned short* hb   = (unsigned short*)alloc((size_t)T_ * H_ * 2);    // A: T x H
    unsigned short* wqbT = (unsigned short*)alloc((size_t)QGW * H_ * 2);   // 8192 x 2048
    unsigned short* wkbT = (unsigned short*)alloc((size_t)KVW * H_ * 2);   // 512 x 2048
    unsigned short* wvbT = (unsigned short*)alloc((size_t)KVW * H_ * 2);
    unsigned short* wobT = (unsigned short*)alloc((size_t)H_ * OW * 2);    // 2048 x 4096
    unsigned short* qg   = (unsigned short*)alloc((size_t)T_ * QGW * 2);   // T x 8192
    unsigned short* kbf  = (unsigned short*)alloc((size_t)T_ * KVW * 2);   // T x 512 (row-major)
    unsigned short* vtb  = (unsigned short*)alloc((size_t)KVW * T_ * 2);   // 512 x T (dim-major)
    unsigned short* ob   = (unsigned short*)alloc((size_t)T_ * OW * 2);    // T x 4096

    // 1) fp32 -> bf16 (A row-major; weights transposed to NxK)
    f32_to_bf16_kernel<<<2048, 256, 0, stream>>>(hidden, hb, (long long)T_ * H_);
    transpose_f32_bf16_kernel<<<dim3(QGW / 32, H_ / 32), dim3(32, 8), 0, stream>>>(Wq, wqbT, H_, QGW);
    transpose_f32_bf16_kernel<<<dim3(KVW / 32, H_ / 32), dim3(32, 8), 0, stream>>>(Wk, wkbT, H_, KVW);
    transpose_f32_bf16_kernel<<<dim3(KVW / 32, H_ / 32), dim3(32, 8), 0, stream>>>(Wv, wvbT, H_, KVW);
    transpose_f32_bf16_kernel<<<dim3(H_ / 32, OW / 32), dim3(32, 8), 0, stream>>>(Wo, wobT, OW, H_);

    // 2) projections (bf16 WMMA)
    gemm_bf16_tn_kernel<<<dim3(QGW / 128, T_ / 64), 128, 0, stream>>>(hb, wqbT, qg,  T_, QGW, H_, 1);
    gemm_bf16_tn_kernel<<<dim3(KVW / 128, T_ / 64), 128, 0, stream>>>(hb, wkbT, kbf, T_, KVW, H_, 1);
    gemm_bf16_tn_kernel<<<dim3(KVW / 128, T_ / 64), 128, 0, stream>>>(hb, wvbT, vtb, T_, KVW, H_, 2); // transposed store

    // 3) RMS norm + RoPE (in place)
    normrope_kernel<<<T_ * NH_,  256, 0, stream>>>(qg,  qw, positions, NH_,  QGW, 2 * HD_);
    normrope_kernel<<<T_ * NKV_, 256, 0, stream>>>(kbf, kw, positions, NKV_, KVW, HD_);

    // 4) flash attention + sigmoid gate
    attn_kernel<<<dim3(T_ / 16, NH_), 128, 0, stream>>>(qg, kbf, vtb, ob);

    // 5) output projection -> fp32
    gemm_bf16_tn_kernel<<<dim3(H_ / 128, T_ / 64), 128, 0, stream>>>(ob, wobT, d_out, T_, H_, OW, 0);
}
